// ProteinDecoder_67997922230500
// MI455X (gfx1250) — matrix operand order repaired and checked by reading, not compiled, not run on hardware
//
#include <hip/hip_runtime.h>
#include <hip/hip_bf16.h>

// ProteinDecoder for MI455X (gfx1250): bf16 WMMA path for the two L^2 GEMMs.
// ND=256, ED=128, H=256, NL=3, NC=24, B=1, L=512.
//
// v2: 32 pairs per block; edge tile converted to bf16 once per block into LDS
// (8x cut in redundant global loads + conversions); 48 wmma per wave.

#define NDd 256
#define EDd 128
#define Hd  256
#define NLd 3
#define NCd 24
#define Ld  512
#define PJT 32          // pairs (j values) per block
#define EPSv 1e-5f

typedef __attribute__((ext_vector_type(16))) __bf16 v16bf;
typedef __attribute__((ext_vector_type(8)))  float  v8f;

union FragB { v16bf v; uint4 u4[2]; unsigned short h[16]; };
union FragC { v8f  v; float f[8]; };

__device__ __forceinline__ float lrelu(float v) { return v > 0.f ? v : 0.01f * v; }

__device__ __forceinline__ unsigned short f2bf(float f) {
    __bf16 b = (__bf16)f;                     // native cvt on gfx1250
    return __builtin_bit_cast(unsigned short, b);
}

// ---------------------------------------------------------------------------
// Pack a KxN (row-major, N==256) fp32 weight matrix into WMMA B-fragment
// order (bf16). Fragment f = kchunk*16 + ntile; per-lane block of 16 halves:
// lane l: N = ntile*16 + (l&15), K = kchunk*32 + (l<16?0:8) + (q<8?q:q+8).
// ---------------------------------------------------------------------------
__global__ void __launch_bounds__(256)
pack_b_kern(const float* __restrict__ src, unsigned short* __restrict__ dst, int Kdim) {
    int gid = blockIdx.x * 256 + threadIdx.x;
    if (gid >= Kdim * Hd) return;
    int f  = gid >> 9;          // fragment index (512 halves each)
    int r  = gid & 511;
    int l  = r >> 4;            // lane 0..31
    int q  = r & 15;            // half within lane
    int c  = f >> 4;            // k-chunk
    int nt = f & 15;            // n-tile
    int K  = c * 32 + ((l & 16) ? 8 : 0) + (q < 8 ? q : q + 8);
    int N  = nt * 16 + (l & 15);
    dst[gid] = f2bf(src[(size_t)K * Hd + N]);
}

// ---------------------------------------------------------------------------
// Per-layer: xi = x@W1i + b1, xj = x@W1j; also zero msum accumulator.
// One block per node i, 256 threads (one per h).
// ---------------------------------------------------------------------------
__global__ void __launch_bounds__(256)
xixj_kern(const float* __restrict__ x, const float* __restrict__ W1k,
          const float* __restrict__ b1k, float* __restrict__ xi,
          float* __restrict__ xj, float* __restrict__ msum) {
    __shared__ float xs[NDd];
    const int i = blockIdx.x, h = threadIdx.x;
    xs[h] = x[(size_t)i * NDd + h];
    __syncthreads();
    float ai = b1k[h], aj = 0.f;
#pragma unroll 4
    for (int d = 0; d < NDd; ++d) {
        float xv = xs[d];
        ai = fmaf(xv, W1k[(size_t)d * Hd + h], ai);
        aj = fmaf(xv, W1k[(size_t)(NDd + d) * Hd + h], aj);
    }
    size_t o = (size_t)i * Hd + h;
    xi[o] = ai; xj[o] = aj; msum[o] = 0.f;
}

// ---------------------------------------------------------------------------
// Fused pair kernel: for one 32-pair tile (fixed i, j=j0..j0+31):
//   h  = lrelu(edge@W1e + xi[i] + xj[j] + b1)      (GEMM1, K=128, bf16 WMMA)
//   m  = lrelu(h@W2 + b2)                           (GEMM2, K=256, bf16 WMMA)
//   msum[i] += sum_j m                              (atomicAdd)
// 256 threads = 8 waves; each wave owns 2 of 16 column tiles of H=256 and
// both 16-pair M-tiles (B fragments reused across M).
// ---------------------------------------------------------------------------
__global__ void __launch_bounds__(256)
pair_kern(const float* __restrict__ edge,
          const float* __restrict__ xib,            // xi + b1 folded
          const float* __restrict__ xj,
          const unsigned short* __restrict__ w1e_pk,
          const unsigned short* __restrict__ w2_pk,
          const float* __restrict__ b2k,
          float* __restrict__ msum) {
    __shared__ float          xjbuf[PJT][Hd];       // 32 KB
    __shared__ float          xibuf[Hd];            //  1 KB
    __shared__ unsigned short ebuf[PJT][EDd];       //  8 KB (bf16 edge tile)
    __shared__ unsigned short hbuf[PJT][Hd];        // 16 KB (bf16 h tile)

    const int tid = threadIdx.x;
    const int t   = blockIdx.x;                     // pair tile 0..8191
    const int i   = t >> 4;                         // L/32 = 16 j-tiles per i
    const int j0  = (t & 15) << 5;

    // Stage xj rows (contiguous in memory) as float4.
    {
        const float4* src  = (const float4*)(xj + (size_t)j0 * Hd);
        float4*       dstv = (float4*)(&xjbuf[0][0]);
        for (int idx = tid; idx < PJT * Hd / 4; idx += 256) dstv[idx] = src[idx];
    }
    xibuf[tid] = xib[(size_t)i * Hd + tid];
    // Stage + convert the 32x128 fp32 edge tile to bf16, once per block.
    {
        const float4* src = (const float4*)(edge + ((size_t)i * Ld + j0) * EDd);
        for (int idx = tid; idx < PJT * EDd / 4; idx += 256) {
            float4 v = src[idx];
            unsigned short* d = &ebuf[0][0] + idx * 4;
            d[0] = f2bf(v.x); d[1] = f2bf(v.y); d[2] = f2bf(v.z); d[3] = f2bf(v.w);
        }
    }
    __syncthreads();

    const int wv   = tid >> 5;
    const int lane = tid & 31;
    const int Ms   = lane & 15;                     // A-fragment row within M-tile
    const int kb   = (lane & 16) ? 8 : 0;           // A-fragment K sub-offset
    const int Mb   = (lane & 16) ? 8 : 0;           // C/D row base
    const int n0   = wv << 1;                       // this wave's 2 col tiles

    // ---------------- GEMM1: edge(32x128) @ W1e(128x256) -------------------
    FragC a00, a01, a10, a11;                       // [mtile][ntile]
#pragma unroll
    for (int r = 0; r < 8; ++r) { a00.f[r] = 0.f; a01.f[r] = 0.f; a10.f[r] = 0.f; a11.f[r] = 0.f; }

#pragma unroll
    for (int c = 0; c < 4; ++c) {
        FragB A0, A1;
        const unsigned short* p0 = &ebuf[Ms][c * 32 + kb];
        const unsigned short* p1 = &ebuf[16 + Ms][c * 32 + kb];
        A0.u4[0] = *(const uint4*)(p0);  A0.u4[1] = *(const uint4*)(p0 + 16);
        A1.u4[0] = *(const uint4*)(p1);  A1.u4[1] = *(const uint4*)(p1 + 16);

        FragB B0, B1;
        const uint4* w0 = (const uint4*)(w1e_pk + (size_t)((c * 16 + n0)     * 32 + lane) * 16);
        const uint4* w1 = (const uint4*)(w1e_pk + (size_t)((c * 16 + n0 + 1) * 32 + lane) * 16);
        B0.u4[0] = w0[0]; B0.u4[1] = w0[1];
        B1.u4[0] = w1[0]; B1.u4[1] = w1[1];

        a00.v = __builtin_amdgcn_wmma_f32_16x16x32_bf16(false, A0.v, false, B0.v, (short)0, a00.v, false, false);
        a01.v = __builtin_amdgcn_wmma_f32_16x16x32_bf16(false, A0.v, false, B1.v, (short)0, a01.v, false, false);
        a10.v = __builtin_amdgcn_wmma_f32_16x16x32_bf16(false, A1.v, false, B0.v, (short)0, a10.v, false, false);
        a11.v = __builtin_amdgcn_wmma_f32_16x16x32_bf16(false, A1.v, false, B1.v, (short)0, a11.v, false, false);
    }

    // Epilogue 1: + xi[i] + b1 + xj[j], leaky-relu, bf16 -> LDS h tile.
#pragma unroll
    for (int mt = 0; mt < 2; ++mt) {
#pragma unroll
        for (int n2 = 0; n2 < 2; ++n2) {
            FragC& ac = mt ? (n2 ? a11 : a10) : (n2 ? a01 : a00);
            int col = (n0 + n2) * 16 + (lane & 15);
            float xiv = xibuf[col];
#pragma unroll
            for (int r = 0; r < 8; ++r) {
                int Mr = mt * 16 + Mb + r;
                float v = ac.f[r] + xiv + xjbuf[Mr][col];
                hbuf[Mr][col] = f2bf(lrelu(v));
            }
        }
    }
    __syncthreads();

    // ---------------- GEMM2: h(32x256) @ W2(256x256) -----------------------
    FragC d00, d01, d10, d11;
#pragma unroll
    for (int r = 0; r < 8; ++r) { d00.f[r] = 0.f; d01.f[r] = 0.f; d10.f[r] = 0.f; d11.f[r] = 0.f; }

#pragma unroll
    for (int c = 0; c < 8; ++c) {
        FragB A0, A1;
        const unsigned short* p0 = &hbuf[Ms][c * 32 + kb];
        const unsigned short* p1 = &hbuf[16 + Ms][c * 32 + kb];
        A0.u4[0] = *(const uint4*)(p0);  A0.u4[1] = *(const uint4*)(p0 + 16);
        A1.u4[0] = *(const uint4*)(p1);  A1.u4[1] = *(const uint4*)(p1 + 16);

        FragB B0, B1;
        const uint4* w0 = (const uint4*)(w2_pk + (size_t)((c * 16 + n0)     * 32 + lane) * 16);
        const uint4* w1 = (const uint4*)(w2_pk + (size_t)((c * 16 + n0 + 1) * 32 + lane) * 16);
        B0.u4[0] = w0[0]; B0.u4[1] = w0[1];
        B1.u4[0] = w1[0]; B1.u4[1] = w1[1];

        d00.v = __builtin_amdgcn_wmma_f32_16x16x32_bf16(false, A0.v, false, B0.v, (short)0, d00.v, false, false);
        d01.v = __builtin_amdgcn_wmma_f32_16x16x32_bf16(false, A0.v, false, B1.v, (short)0, d01.v, false, false);
        d10.v = __builtin_amdgcn_wmma_f32_16x16x32_bf16(false, A1.v, false, B0.v, (short)0, d10.v, false, false);
        d11.v = __builtin_amdgcn_wmma_f32_16x16x32_bf16(false, A1.v, false, B1.v, (short)0, d11.v, false, false);
    }

    // Epilogue 2: +b2, leaky-relu, reduce over the 32 j's, atomicAdd to msum.
#pragma unroll
    for (int n2 = 0; n2 < 2; ++n2) {
        int col = (n0 + n2) * 16 + (lane & 15);
        float b2v = b2k[col];
        float s = 0.f;
        FragC& e0 = n2 ? d01 : d00;
        FragC& e1 = n2 ? d11 : d10;
#pragma unroll
        for (int r = 0; r < 8; ++r) {
            s += lrelu(e0.f[r] + b2v);
            s += lrelu(e1.f[r] + b2v);
        }
        s += __shfl_xor(s, 16, 32);                 // combine M=0..7 and M=8..15 halves
        if (lane < 16) atomicAdd(&msum[(size_t)i * Hd + col], s);
    }
}

// ---------------------------------------------------------------------------
// Per-layer tail: m_i = lrelu(msum@Wu + bu); y = x + m_i; layernorm -> x_out.
// One block per node i, 256 threads.
// ---------------------------------------------------------------------------
__global__ void __launch_bounds__(256)
update_kern(const float* __restrict__ xin, const float* __restrict__ msum,
            const float* __restrict__ Wuk, const float* __restrict__ buk,
            const float* __restrict__ gk, const float* __restrict__ bk,
            float* __restrict__ xout) {
    __shared__ float ms[Hd];
    __shared__ float red[Hd];
    const int i = blockIdx.x, h = threadIdx.x;
    ms[h] = msum[(size_t)i * Hd + h];
    __syncthreads();
    float a = buk[h];
#pragma unroll 4
    for (int d = 0; d < Hd; ++d) a = fmaf(ms[d], Wuk[(size_t)d * Hd + h], a);
    float y = xin[(size_t)i * Hd + h] + lrelu(a);

    red[h] = y; __syncthreads();
    for (int s = 128; s > 0; s >>= 1) { if (h < s) red[h] += red[h + s]; __syncthreads(); }
    float mu = red[0] * (1.f / Hd);
    __syncthreads();
    float dv = y - mu;
    red[h] = dv * dv; __syncthreads();
    for (int s = 128; s > 0; s >>= 1) { if (h < s) red[h] += red[h + s]; __syncthreads(); }
    float var = red[0] * (1.f / Hd);
    xout[(size_t)i * Hd + h] = dv * rsqrtf(var + EPSv) * gk[h] + bk[h];
}

// ---------------------------------------------------------------------------
// Final projection: out = x @ Wo + bo  (H=256 -> NC=24).
// ---------------------------------------------------------------------------
__global__ void __launch_bounds__(64)
out_kern(const float* __restrict__ x, const float* __restrict__ Wo,
         const float* __restrict__ bo, float* __restrict__ out) {
    __shared__ float xs[Hd];
    const int i = blockIdx.x, tl = threadIdx.x;
    for (int idx = tl; idx < Hd; idx += 64) xs[idx] = x[(size_t)i * Hd + idx];
    __syncthreads();
    if (tl < NCd) {
        float a = bo[tl];
#pragma unroll 4
        for (int hh = 0; hh < Hd; ++hh) a = fmaf(xs[hh], Wo[(size_t)hh * NCd + tl], a);
        out[(size_t)i * NCd + tl] = a;
    }
}

extern "C" void kernel_launch(void* const* d_in, const int* in_sizes, int n_in,
                              void* d_out, int out_size, void* d_ws, size_t ws_size,
                              hipStream_t stream) {
    (void)in_sizes; (void)n_in; (void)out_size; (void)ws_size;
    const float* binder_node = (const float*)d_in[0];   // (1, 512, 256)
    const float* binder_edge = (const float*)d_in[1];   // (1, 512, 512, 128)
    const float* W1    = (const float*)d_in[2];         // (3, 640, 256)
    const float* b1    = (const float*)d_in[3];         // (3, 256)
    const float* W2    = (const float*)d_in[4];         // (3, 256, 256)
    const float* b2    = (const float*)d_in[5];         // (3, 256)
    const float* Wu    = (const float*)d_in[6];         // (3, 256, 256)
    const float* bu    = (const float*)d_in[7];         // (3, 256)
    const float* gamma = (const float*)d_in[8];         // (3, 256)
    const float* beta  = (const float*)d_in[9];         // (3, 256)
    const float* Wo    = (const float*)d_in[10];        // (256, 24)
    const float* bo    = (const float*)d_in[11];        // (24,)

    // Workspace carve-up (~2.7 MB total).
    float* xi   = (float*)d_ws;                     // 512*256 f32
    float* xjw  = xi   + (size_t)Ld * Hd;
    float* msum = xjw  + (size_t)Ld * Hd;
    float* xcur = msum + (size_t)Ld * Hd;
    unsigned short* w1e_pk = (unsigned short*)(xcur + (size_t)Ld * Hd);
    unsigned short* w2_pk  = w1e_pk + (size_t)NLd * EDd * Hd;

    // Pack weights into B-fragment order (once per call; deterministic).
    for (int k = 0; k < NLd; ++k) {
        pack_b_kern<<<(EDd * Hd + 255) / 256, 256, 0, stream>>>(
            W1 + ((size_t)k * (2 * NDd + EDd) + 2 * NDd) * Hd,
            w1e_pk + (size_t)k * EDd * Hd, EDd);
        pack_b_kern<<<(Hd * Hd + 255) / 256, 256, 0, stream>>>(
            W2 + (size_t)k * Hd * Hd,
            w2_pk + (size_t)k * Hd * Hd, Hd);
    }

    const float* xin = binder_node;
    for (int k = 0; k < NLd; ++k) {
        xixj_kern<<<Ld, 256, 0, stream>>>(
            xin, W1 + (size_t)k * (2 * NDd + EDd) * Hd, b1 + (size_t)k * Hd,
            xi, xjw, msum);
        pair_kern<<<(Ld * Ld) / PJT, 256, 0, stream>>>(
            binder_edge, xi, xjw,
            w1e_pk + (size_t)k * EDd * Hd,
            w2_pk + (size_t)k * Hd * Hd,
            b2 + (size_t)k * Hd, msum);
        update_kern<<<Ld, 256, 0, stream>>>(
            xin, msum, Wu + (size_t)k * Hd * Hd, bu + (size_t)k * Hd,
            gamma + (size_t)k * Hd, beta + (size_t)k * Hd, xcur);
        xin = xcur;
    }
    out_kern<<<Ld, 64, 0, stream>>>(xcur, Wo, bo, (float*)d_out);
}
